// Encoder_33225867002101
// MI455X (gfx1250) — matrix operand order repaired and checked by reading, not compiled, not run on hardware
//
#include <hip/hip_runtime.h>
#include <stdint.h>

// ---------------------------------------------------------------------------
// Types for WMMA fragments (gfx1250, wave32).
// ---------------------------------------------------------------------------
typedef __attribute__((ext_vector_type(16))) __bf16 v16bf;
typedef __attribute__((ext_vector_type(8)))  float  v8f;
typedef __attribute__((ext_vector_type(4)))  unsigned int v4u;  // 16-byte load unit

union ABFrag { v4u q[2]; v16bf v; };

__device__ __forceinline__ unsigned short f2bf(float f) {
  unsigned int u = __float_as_uint(f);
  u += 0x7fffu + ((u >> 16) & 1u);           // round-to-nearest-even
  return (unsigned short)(u >> 16);
}
__device__ __forceinline__ float sigmoid_fast(float x) {
  return 1.0f / (1.0f + __expf(-x));
}
__device__ __forceinline__ float softplus_fast(float x) {
  return (x > 15.0f) ? x : __logf(1.0f + __expf(x));
}

// ---------------------------------------------------------------------------
// Prep: fp32 [K][N] row-major  ->  bf16 [N][K] row-major (transpose+convert).
// Puts the WMMA B operand in "N-major, K-contiguous" form so each lane's
// 16-element K run is two contiguous 16B loads.
// ---------------------------------------------------------------------------
__global__ void cvt_transpose(const float* __restrict__ src,
                              unsigned short* __restrict__ dst,
                              int K, int N) {
  long long idx = (long long)blockIdx.x * blockDim.x + threadIdx.x;
  long long total = (long long)K * N;
  if (idx >= total) return;
  int n = (int)(idx / K);
  int k = (int)(idx % K);
  dst[(long long)n * K + k] = f2bf(src[(long long)k * N + n]);
}

__global__ void cvt_copy(const float* __restrict__ src,
                         unsigned short* __restrict__ dst, long long total) {
  long long idx = (long long)blockIdx.x * blockDim.x + threadIdx.x;
  if (idx < total) dst[idx] = f2bf(src[idx]);
}

__global__ void zero_state(float* __restrict__ c,
                           unsigned short* __restrict__ h0, int total) {
  int idx = blockIdx.x * blockDim.x + threadIdx.x;
  if (idx < total) { c[idx] = 0.0f; h0[idx] = 0; }
}

// ---------------------------------------------------------------------------
// One LSTM timestep.  Grid (8, 32), block 256 (8 waves).
// Workgroup tile: 16 rows (M) x 32 h-columns, all 4 gates.
//   wave w: gate = w&3, column-halftile = w>>2 -> one 16x16 WMMA output tile,
//   K = 256 (x@W) + 1024 (h@U), bf16 inputs, fp32 accumulation.
// A rows [x_t | h] staged once in LDS (40 KB) and reused by all 8 waves.
// Gate tiles exchanged through LDS for the c/h elementwise update.
// ---------------------------------------------------------------------------
__global__ __launch_bounds__(256) void lstm_step(
    const unsigned short* __restrict__ xbf,   // [B=128][T=512][D=256] bf16
    const unsigned short* __restrict__ Wt,    // [4096][256]  bf16 (W^T)
    const unsigned short* __restrict__ Ut,    // [4096][1024] bf16 (U^T)
    const float*          __restrict__ bias,  // [4096]
    const unsigned short* __restrict__ hin,   // [128][1024] bf16
    unsigned short*       __restrict__ hout,  // [128][1024] bf16
    float*                __restrict__ c,     // [128][1024] fp32 (in-place)
    int t) {
  __shared__ __align__(16) unsigned short lds_a[16 * 1280];  // [x(256) | h(1024)]
  __shared__ float lds_g[4 * 32 * 16];                       // gate,col,row

  const int tid = threadIdx.x;
  const int r0  = blockIdx.x * 16;    // M base
  const int n0  = blockIdx.y * 32;    // h-column base

  // ---- stage A = [x_t(16x256) | h(16x1024)] into LDS: 2560 x 16B chunks ----
  #pragma unroll
  for (int i = 0; i < 10; ++i) {
    int cid = tid + i * 256;          // 0..2559
    int m   = cid / 160;              // row 0..15
    int off = (cid % 160) * 8;        // bf16 element offset 0..1272
    v4u v;
    if (off < 256)
      v = *(const v4u*)(xbf + (size_t)(r0 + m) * (512 * 256) +
                        (size_t)t * 256 + off);
    else
      v = *(const v4u*)(hin + (size_t)(r0 + m) * 1024 + (off - 256));
    *(v4u*)&lds_a[m * 1280 + off] = v;
  }
  __syncthreads();

  const int lane = tid & 31;
  const int l15  = lane & 15;
  const int lhi  = lane >> 4;
  const int wv   = tid >> 5;
  const int g    = wv & 3;            // gate i/f/g/o
  const int sel  = wv >> 2;           // which 16-col half of the 32-col block
  const int gcol = g * 1024 + n0 + sel * 16 + l15;  // column in 4H space

  v8f acc = {0.f, 0.f, 0.f, 0.f, 0.f, 0.f, 0.f, 0.f};

  // ---- x_t @ W : K = 256 ----
  const unsigned short* bpW = Wt + (size_t)gcol * 256 + lhi * 16;
  #pragma unroll
  for (int k0 = 0; k0 < 256; k0 += 32) {
    ABFrag a, b;
    const v4u* pa = (const v4u*)&lds_a[l15 * 1280 + k0 + lhi * 16];
    a.q[0] = pa[0]; a.q[1] = pa[1];          // K: [k0+16h..+15] per lane half
    const v4u* pb = (const v4u*)(bpW + k0);
    b.q[0] = pb[0]; b.q[1] = pb[1];          // 16 contiguous K per lane
    acc = __builtin_amdgcn_wmma_f32_16x16x32_bf16(false, a.v, false, b.v,
                                                  (short)0, acc, false, false);
  }
  // ---- h @ U : K = 1024 ----
  const unsigned short* bpU = Ut + (size_t)gcol * 1024 + lhi * 16;
  #pragma unroll 4
  for (int k0 = 0; k0 < 1024; k0 += 32) {
    ABFrag a, b;
    const v4u* pa = (const v4u*)&lds_a[l15 * 1280 + 256 + k0 + lhi * 16];
    a.q[0] = pa[0]; a.q[1] = pa[1];
    const v4u* pb = (const v4u*)(bpU + k0);
    b.q[0] = pb[0]; b.q[1] = pb[1];
    acc = __builtin_amdgcn_wmma_f32_16x16x32_bf16(false, a.v, false, b.v,
                                                  (short)0, acc, false, false);
  }

  // ---- publish gate tile (+bias) to LDS ----
  const float bb = bias[gcol];
  #pragma unroll
  for (int j = 0; j < 8; ++j)
    lds_g[(g * 32 + sel * 16 + l15) * 16 + lhi * 8 + j] = acc[j] + bb;
  __syncthreads();

  // ---- elementwise LSTM update: 512 elements over 256 threads ----
  #pragma unroll
  for (int e = tid; e < 512; e += 256) {
    int m = e & 15, col = e >> 4;
    float iv = sigmoid_fast (lds_g[(0 * 32 + col) * 16 + m]);
    float fv = sigmoid_fast (lds_g[(1 * 32 + col) * 16 + m]);
    float gv = softplus_fast(lds_g[(2 * 32 + col) * 16 + m]);
    float ov = sigmoid_fast (lds_g[(3 * 32 + col) * 16 + m]);
    size_t idx = (size_t)(r0 + m) * 1024 + n0 + col;
    float cn = fv * c[idx] + iv * gv;
    c[idx] = cn;
    hout[idx] = f2bf(ov * softplus_fast(cn));
  }
}

// ---------------------------------------------------------------------------
// Projection: mu = h@Wm + bm, logvar = h@Wv + bv, z = mu + eps*exp(0.5*lv).
// Grid (8,8), block 32 (one wave -> one 16x16 tile, two WMMA accumulators
// with identical C/D layout so z is computed entirely in registers).
// ---------------------------------------------------------------------------
__global__ __launch_bounds__(32) void proj_kernel(
    const unsigned short* __restrict__ hbf,   // [128][1024] bf16
    const unsigned short* __restrict__ Wmt,   // [128][1024] bf16 (Wm^T)
    const unsigned short* __restrict__ Wvt,   // [128][1024] bf16 (Wv^T)
    const float* __restrict__ bm,
    const float* __restrict__ bv,
    const float* __restrict__ eps,            // [128][128]
    float* __restrict__ out) {                // mu | logvar | z (each 128x128)
  const int lane = threadIdx.x & 31;
  const int l15  = lane & 15;
  const int lhi  = lane >> 4;
  const int r0   = blockIdx.x * 16;
  const int n0   = blockIdx.y * 16;

  v8f am = {0.f, 0.f, 0.f, 0.f, 0.f, 0.f, 0.f, 0.f};
  v8f av = {0.f, 0.f, 0.f, 0.f, 0.f, 0.f, 0.f, 0.f};

  const unsigned short* pa0 = hbf + (size_t)(r0 + l15) * 1024 + lhi * 16;
  const unsigned short* pbm = Wmt + (size_t)(n0 + l15) * 1024 + lhi * 16;
  const unsigned short* pbv = Wvt + (size_t)(n0 + l15) * 1024 + lhi * 16;

  #pragma unroll 4
  for (int k0 = 0; k0 < 1024; k0 += 32) {
    ABFrag a, b1, b2;
    const v4u* pa = (const v4u*)(pa0 + k0);
    a.q[0] = pa[0]; a.q[1] = pa[1];
    const v4u* p1 = (const v4u*)(pbm + k0);
    b1.q[0] = p1[0]; b1.q[1] = p1[1];
    const v4u* p2 = (const v4u*)(pbv + k0);
    b2.q[0] = p2[0]; b2.q[1] = p2[1];
    am = __builtin_amdgcn_wmma_f32_16x16x32_bf16(false, a.v, false, b1.v,
                                                 (short)0, am, false, false);
    av = __builtin_amdgcn_wmma_f32_16x16x32_bf16(false, a.v, false, b2.v,
                                                 (short)0, av, false, false);
  }

  const int n   = n0 + l15;
  const float bmv = bm[n];
  const float bvv = bv[n];
  #pragma unroll
  for (int j = 0; j < 8; ++j) {
    int m = r0 + lhi * 8 + j;
    float mu = am[j] + bmv;
    float lv = av[j] + bvv;
    float z  = mu + eps[(size_t)m * 128 + n] * __expf(0.5f * lv);
    out[(size_t)m * 128 + n]           = mu;
    out[16384 + (size_t)m * 128 + n]   = lv;
    out[32768 + (size_t)m * 128 + n]   = z;
  }
}

// ---------------------------------------------------------------------------
// Host-side launcher.
// ---------------------------------------------------------------------------
extern "C" void kernel_launch(void* const* d_in, const int* in_sizes, int n_in,
                              void* d_out, int out_size, void* d_ws,
                              size_t ws_size, hipStream_t stream) {
  const float* x   = (const float*)d_in[0];   // [128,512,256]
  const float* W   = (const float*)d_in[1];   // [256,4096]
  const float* U   = (const float*)d_in[2];   // [1024,4096]
  const float* b   = (const float*)d_in[3];   // [4096]
  const float* Wm  = (const float*)d_in[4];   // [1024,128]
  const float* bm  = (const float*)d_in[5];   // [128]
  const float* Wv  = (const float*)d_in[6];   // [1024,128]
  const float* bv  = (const float*)d_in[7];   // [128]
  const float* eps = (const float*)d_in[8];   // [128,128]
  float* out = (float*)d_out;

  // Workspace layout (bytes, all 256B aligned).
  const size_t OFF_WT  = 0;            // 4096*256*2   =  2,097,152
  const size_t OFF_UT  = 2097152;      // 4096*1024*2  =  8,388,608
  const size_t OFF_WMT = 10485760;     // 128*1024*2   =    262,144
  const size_t OFF_WVT = 10747904;     // 128*1024*2   =    262,144
  const size_t OFF_XBF = 11010048;     // 128*512*256*2= 33,554,432
  const size_t OFF_H0  = 44564480;     // 128*1024*2   =    262,144
  const size_t OFF_H1  = 44826624;     // 128*1024*2   =    262,144
  const size_t OFF_C   = 45088768;     // 128*1024*4   =    524,288
  const size_t NEED    = 45613056;
  if (ws_size < NEED) return;

  char* ws = (char*)d_ws;
  unsigned short* Wt   = (unsigned short*)(ws + OFF_WT);
  unsigned short* Ut   = (unsigned short*)(ws + OFF_UT);
  unsigned short* Wmt  = (unsigned short*)(ws + OFF_WMT);
  unsigned short* Wvt  = (unsigned short*)(ws + OFF_WVT);
  unsigned short* xbf  = (unsigned short*)(ws + OFF_XBF);
  unsigned short* hb[2] = {(unsigned short*)(ws + OFF_H0),
                           (unsigned short*)(ws + OFF_H1)};
  float* c = (float*)(ws + OFF_C);

  // One-time conversions (amortized over 512 recurrent steps).
  cvt_transpose<<<(4096 * 256 + 255) / 256, 256, 0, stream>>>(W, Wt, 256, 4096);
  cvt_transpose<<<(4096 * 1024 + 255) / 256, 256, 0, stream>>>(U, Ut, 1024, 4096);
  cvt_transpose<<<(128 * 1024 + 255) / 256, 256, 0, stream>>>(Wm, Wmt, 1024, 128);
  cvt_transpose<<<(128 * 1024 + 255) / 256, 256, 0, stream>>>(Wv, Wvt, 1024, 128);
  cvt_copy<<<(16777216 + 255) / 256, 256, 0, stream>>>(x, xbf, 16777216LL);
  zero_state<<<(131072 + 255) / 256, 256, 0, stream>>>(c, hb[0], 131072);

  // Sequential recurrence: 512 graph nodes, h double-buffered (bf16).
  for (int t = 0; t < 512; ++t) {
    lstm_step<<<dim3(8, 32), 256, 0, stream>>>(
        xbf, Wt, Ut, b, hb[t & 1], hb[(t + 1) & 1], c, t);
  }

  // Final VAE head; after t=511 the latest h lives in hb[0].
  proj_kernel<<<dim3(8, 8), 32, 0, stream>>>(hb[0], Wmt, Wvt, bm, bv, eps, out);
}
